// Wasserstein2Loss_71090298683881
// MI455X (gfx1250) — matrix operand order repaired and checked by reading, not compiled, not run on hardware
//
#include <hip/hip_runtime.h>
#include <hip/hip_bf16.h>

typedef _Float16 v8h  __attribute__((ext_vector_type(8)));
typedef _Float16 v16h __attribute__((ext_vector_type(16)));
typedef float    v8f  __attribute__((ext_vector_type(8)));

#define NN   192
#define VOL  (192*192*192)      // 7077888
#define ROWS (192*192)          // 36864 rows per axis pass
#define WEPS 1e-3f
#define MAXITER 20
#define TINYF 1.2e-38f          // smallest normal-ish: keep v_log_f32 out of denorms

// Raw hardware transcendentals (hazards tracked by compiler, unlike inline asm)
__device__ __forceinline__ float fexp(float x) {
#if __has_builtin(__builtin_amdgcn_exp2f)
    return __builtin_amdgcn_exp2f(x * 1.4426950408889634f);   // v_exp_f32
#else
    return __expf(x);
#endif
}
__device__ __forceinline__ float flog(float x) {
#if __has_builtin(__builtin_amdgcn_logf)
    return __builtin_amdgcn_logf(x) * 0.6931471805599453f;    // v_log_f32
#else
    return __logf(x);
#endif
}

// ---------------- Gaussian log-kernel table, f16 ----------------
__global__ void build_G(_Float16* __restrict__ Gh) {
    int idx = blockIdx.x * blockDim.x + threadIdx.x;
    if (idx >= NN * NN) return;
    int i = idx / NN, j = idx % NN;
    const float b = WEPS * (float)NN * (float)NN;   // eps * n^2
    float d = (float)(i - j);
    Gh[idx] = (_Float16)__expf(-(d * d) / b);
}

// ---------------- la = log(max(a,1e-20)); phi = 0 ----------------
__global__ void init_la(const float* __restrict__ a0, const float* __restrict__ a1,
                        float* __restrict__ la0, float* __restrict__ la1,
                        float* __restrict__ phi0, float* __restrict__ phi1) {
    int i = blockIdx.x * blockDim.x + threadIdx.x;
    if (i >= VOL) return;
    la0[i]  = __logf(fmaxf(a0[i], 1e-20f));
    la1[i]  = __logf(fmaxf(a1[i], 1e-20f));
    phi0[i] = 0.f;
    phi1[i] = 0.f;
}

__global__ void sub_kernel(const float* __restrict__ la, const float* __restrict__ t,
                           float* __restrict__ phi) {
    int i = blockIdx.x * blockDim.x + threadIdx.x;
    if (i < VOL) phi[i] = la[i] - t[i];
}

// ---------------- fused LSE blur along one axis ----------------
// 384 threads = 12 waves; block owns 64 rows (4 M-tiles of 16).
// Each wave owns one 16-wide N-tile: its 6 B fragments (f16 G, symmetric so
// column-major B == row-major G reads) stay in registers for the whole kernel.
// A fragments are double-buffered across M-tiles to hide LDS latency.
// Row r: base = (r/INNER)*HI + (r%INNER)*LO, element = base + k*KS.
template <int INNER, long HI, long LO, long KS>
__global__ __launch_bounds__(384)
void blur_axis(const float* __restrict__ src, float* __restrict__ dst,
               const _Float16* __restrict__ Gh) {
    constexpr bool CONTIG = (KS == 1);
    __shared__ __align__(16) _Float16 At[64][200]; // 64x192 exp tile, 16B-aligned rows
    __shared__ float red[64][6];
    __shared__ float smax[64];
    __shared__ __align__(16) float Ot[12][16][16]; // per-wave C transpose buffer

    const int t      = threadIdx.x;
    const int r0     = blockIdx.x * 64;
    const int hiIdx  = r0 / INNER;
    const int loBase = r0 % INNER;

    const int w     = t >> 5;
    const int lane  = t & 31;
    const int mlane = lane & 15;
    const int hl    = lane >> 4;
    const int n     = w * 16 + mlane;          // this wave's output column

    // ---- B fragments: loaded once, overlap with staging below ----
    v16h b[6];
#pragma unroll
    for (int kc = 0; kc < 6; ++kc) {
        v8h blo = *(const v8h*)&Gh[n * NN + kc * 32 + 8 * hl];
        v8h bhi = *(const v8h*)&Gh[n * NN + kc * 32 + 16 + 8 * hl];
        b[kc] = __builtin_shufflevector(blo, bhi,
                   0,1,2,3,4,5,6,7,8,9,10,11,12,13,14,15);
    }

    // ---- stage 1: load 64x192 rows, per-row max, exp -> f16 LDS ----
    int m, kp;
    float v[32];
    if constexpr (CONTIG) {
        m = t / 6; kp = t % 6;                 // 32 consecutive k per thread (float4)
        const float4* p = (const float4*)(src + (long)hiIdx * HI
                                          + (long)(loBase + m) * LO + kp * 32);
#pragma unroll
        for (int j = 0; j < 8; ++j) {
            float4 q = p[j];
            v[4*j+0] = q.x; v[4*j+1] = q.y; v[4*j+2] = q.z; v[4*j+3] = q.w;
        }
    } else {
        m = t & 63; kp = t >> 6;               // consecutive threads = consecutive rows
        const long base = (long)hiIdx * HI + (long)(loBase + m) * LO;
#pragma unroll
        for (int j = 0; j < 32; ++j)
            v[j] = src[base + (long)(kp * 32 + j) * KS];
    }
    float mx = v[0];
#pragma unroll
    for (int j = 1; j < 32; ++j) mx = fmaxf(mx, v[j]);
    red[m][kp] = mx;
    __syncthreads();
    float s = red[m][0];
#pragma unroll
    for (int j = 1; j < 6; ++j) s = fmaxf(s, red[m][j]);
    if (kp == 0) smax[m] = s;
#pragma unroll
    for (int j = 0; j < 32; ++j)
        At[m][kp * 32 + j] = (_Float16)fexp(v[j] - s);
    __syncthreads();

    // ---- stage 2: 4 M-tiles, A fragments double-buffered ----
    // A layout: lane(hl,M) holds K = kb+8*hl+0..7 and kb+16+8*hl+0..7
    auto loadA = [&](int Mb, v16h* a) {
#pragma unroll
        for (int kc = 0; kc < 6; ++kc) {
            int kb = kc * 32;
            v8h lo8 = *(const v8h*)&At[Mb + mlane][kb + 8 * hl];
            v8h hi8 = *(const v8h*)&At[Mb + mlane][kb + 16 + 8 * hl];
            a[kc] = __builtin_shufflevector(lo8, hi8,
                       0,1,2,3,4,5,6,7,8,9,10,11,12,13,14,15);
        }
    };

    v16h abuf[2][6];
    loadA(0, abuf[0]);
#pragma unroll
    for (int mt = 0; mt < 4; ++mt) {
        const int Mb = mt * 16;
        if (mt < 3) loadA(Mb + 16, abuf[(mt + 1) & 1]);   // prefetch next tile
        const v16h* a = abuf[mt & 1];

        v8f c = {};
#pragma unroll
        for (int kc = 0; kc < 6; ++kc)
            c = __builtin_amdgcn_wmma_f32_16x16x32_f16(
                    false, a[kc], false, b[kc], (short)0, c, false, false);

        if constexpr (CONTIG) {
            // direct stores: lanes span n (stride 1) -> coalesced per half-wave
#pragma unroll
            for (int r = 0; r < 8; ++r) {
                int M = Mb + r + 8 * hl;
                float val = flog(fmaxf(c[r], TINYF)) + smax[M];
                dst[(long)hiIdx * HI + (long)(loBase + M) * LO + (long)n] = val;
            }
        } else {
            // transpose C through LDS, then float4 stores with unit row stride
#pragma unroll
            for (int r = 0; r < 8; ++r) {
                int M = Mb + r + 8 * hl;
                float val = flog(fmaxf(c[r], TINYF)) + smax[M];
                Ot[w][mlane][r + 8 * hl] = val;
            }
            __syncthreads();
#pragma unroll
            for (int i = 0; i < 2; ++i) {
                int nl = i * 8 + (lane >> 2);
                int Mq = (lane & 3) * 4;
                float4 q = *(const float4*)&Ot[w][nl][Mq];
                long addr = (long)hiIdx * HI + (long)(loBase + Mb + Mq) * LO
                          + (long)(w * 16 + nl) * KS;     // LO == 1 here
                *(float4*)&dst[addr] = q;
            }
            __syncthreads();
        }
    }
}

// ---------------- loss reduction ----------------
__global__ void zero_out(float* o) {
    if (blockIdx.x == 0 && threadIdx.x == 0) o[0] = 0.f;
}

__global__ __launch_bounds__(256)
void loss_reduce(const float* __restrict__ phi0, const float* __restrict__ phi1,
                 const float* __restrict__ bl1,
                 const float* __restrict__ a0,  const float* __restrict__ a1,
                 float* __restrict__ out) {
    __shared__ float sh[256];
    float acc = 0.f;
    for (long i = (long)blockIdx.x * blockDim.x + threadIdx.x; i < VOL;
         i += (long)gridDim.x * blockDim.x) {
        acc += phi0[i] * a0[i] + phi1[i] * a1[i] - fexp(phi0[i] + bl1[i]);
    }
    sh[threadIdx.x] = acc;
    __syncthreads();
    for (int s = 128; s > 0; s >>= 1) {
        if (threadIdx.x < s) sh[threadIdx.x] += sh[threadIdx.x + s];
        __syncthreads();
    }
    if (threadIdx.x == 0) atomicAdd(out, WEPS * sh[0]);
}

// ---------------- host driver ----------------
extern "C" void kernel_launch(void* const* d_in, const int* in_sizes, int n_in,
                              void* d_out, int out_size, void* d_ws, size_t ws_size,
                              hipStream_t stream) {
    const float* a0 = (const float*)d_in[0];
    const float* a1 = (const float*)d_in[1];
    float* out = (float*)d_out;

    char* ws = (char*)d_ws;
    _Float16* Gh = (_Float16*)ws;
    size_t o = (size_t)NN * NN * sizeof(_Float16);   // 73728, 256B-aligned
    float* la0  = (float*)(ws + o); o += (size_t)VOL * 4;
    float* la1  = (float*)(ws + o); o += (size_t)VOL * 4;
    float* phi0 = (float*)(ws + o); o += (size_t)VOL * 4;
    float* phi1 = (float*)(ws + o); o += (size_t)VOL * 4;
    float* tmp  = (float*)(ws + o); o += (size_t)VOL * 4;

    const int EB = (VOL + 255) / 256;
    const int BG = ROWS / 64;                        // 576 blocks per axis pass
    build_G<<<(NN * NN + 255) / 256, 256, 0, stream>>>(Gh);
    init_la<<<EB, 256, 0, stream>>>(a0, a1, la0, la1, phi0, phi1);

    auto blur3 = [&](const float* s, float* d) {
        // axis 0: rows (y,x) base=y*192+x, k stride N^2  (out-of-place)
        blur_axis<192, 192L, 1L, (long)ROWS><<<BG, 384, 0, stream>>>(s, d, Gh);
        // axis 1: rows (z,x) base=z*N^2+x, k stride 192  (in-place)
        blur_axis<192, (long)ROWS, 1L, 192L><<<BG, 384, 0, stream>>>(d, d, Gh);
        // axis 2: rows r,    base=r*192,   k stride 1    (in-place)
        blur_axis<ROWS, 0L, 192L, 1L><<<BG, 384, 0, stream>>>(d, d, Gh);
    };

    for (int it = 0; it < MAXITER; ++it) {
        blur3(phi1, tmp);
        sub_kernel<<<EB, 256, 0, stream>>>(la0, tmp, phi0);
        blur3(phi0, tmp);
        sub_kernel<<<EB, 256, 0, stream>>>(la1, tmp, phi1);
    }

    blur3(phi1, tmp);
    zero_out<<<1, 64, 0, stream>>>(out);
    loss_reduce<<<2048, 256, 0, stream>>>(phi0, phi1, tmp, a0, a1, out);
}
